// PrototypeBase_79448305041748
// MI455X (gfx1250) — matrix operand order repaired and checked by reading, not compiled, not run on hardware
//
#include <hip/hip_runtime.h>
#include <hip/hip_bf16.h>
#include <float.h>

// MI455X / gfx1250 (wave32, WMMA). Prototype loss:
//   dist = sqrt(max(z2[n] + p2[p] - 2*z@p^T, 1e-12)); out = .05*mean(rowmin) + .05*mean(colmin)
// GEMM done as bf16 split (hi/lo): hh + lh + hl via v_wmma_f32_16x16x32_bf16 (f32 accum).
// z2/p2 computed in exact f32 and added in the epilogue.

typedef __attribute__((ext_vector_type(16))) __bf16 v16bf;
typedef __attribute__((ext_vector_type(8)))  __bf16 v8bf;
typedef __attribute__((ext_vector_type(8)))  float  v8f;

#define N_TOT 65536
#define P_TOT 512
#define D_TOT 256
#define M_TILE 64
#define LDS_STRIDE 264  // 256 + 8 bf16 pad (16B): 528B row stride == 16 mod 256 -> bank spread

// Workspace layout (bytes). Requires ws_size >= ~520 KB.
static constexpr size_t WS_ROWSUM = 0;                                   // float
static constexpr size_t WS_COLMIN = 64;                                  // u32[512]
static constexpr size_t WS_P2     = 2112;                                // float[512]
static constexpr size_t WS_PHI    = 4224;                                // bf16[512*256]
static constexpr size_t WS_PLO    = WS_PHI + (size_t)P_TOT * D_TOT * 2;  // bf16[512*256]

__global__ void pb_init_kernel(float* __restrict__ rowsum, unsigned* __restrict__ colmin) {
  const int t = threadIdx.x;
  if (t == 0) *rowsum = 0.0f;
  colmin[t]       = 0x7F7FFFFFu;  // +FLT_MAX bit pattern
  colmin[t + 256] = 0x7F7FFFFFu;
}

// One wave per prototype row: bf16 hi/lo decomposition + exact f32 squared norm.
__global__ void pb_proto_prep_kernel(const float* __restrict__ protos,
                                     __bf16* __restrict__ phi, __bf16* __restrict__ plo,
                                     float* __restrict__ p2) {
  const int w = threadIdx.x >> 5, lane = threadIdx.x & 31;
  const int pr = blockIdx.x * 8 + w;
  const float* row = protos + (size_t)pr * D_TOT;
  const int c0 = lane * 8;
  float s = 0.0f;
#pragma unroll
  for (int ii = 0; ii < 8; ++ii) {
    float v = row[c0 + ii];
    __bf16 h = (__bf16)v;
    __bf16 l = (__bf16)(v - (float)h);
    phi[(size_t)pr * D_TOT + c0 + ii] = h;
    plo[(size_t)pr * D_TOT + c0 + ii] = l;
    s += v * v;
  }
  s += __shfl_xor(s, 1);  s += __shfl_xor(s, 2);  s += __shfl_xor(s, 4);
  s += __shfl_xor(s, 8);  s += __shfl_xor(s, 16);
  if (lane == 0) p2[pr] = s;
}

__device__ __forceinline__ v16bf pb_frag16(const __bf16* p) {
  // A-matrix per-lane layout: elems 0..7 at +0, elems 8..15 at +16 (bf16 units)
  v8bf a = *reinterpret_cast<const v8bf*>(p);
  v8bf b = *reinterpret_cast<const v8bf*>(p + 16);
  return __builtin_shufflevector(a, b, 0, 1, 2, 3, 4, 5, 6, 7, 8, 9, 10, 11, 12, 13, 14, 15);
}

__global__ void __launch_bounds__(256)
pb_dist_kernel(const float* __restrict__ z,
               const __bf16* __restrict__ phi, const __bf16* __restrict__ plo,
               const float* __restrict__ p2,
               float* __restrict__ rowsum, unsigned* __restrict__ colmin) {
  extern __shared__ __align__(32) char smem[];
  __bf16*   s_zhi    = (__bf16*)smem;                       // [64][264]
  __bf16*   s_zlo    = s_zhi + M_TILE * LDS_STRIDE;         // [64][264]
  float*    s_zsq    = (float*)(s_zlo + M_TILE * LDS_STRIDE);
  unsigned* s_rowmin = (unsigned*)(s_zsq + M_TILE);

  const int t  = threadIdx.x;
  const int n0 = blockIdx.x * M_TILE;

  // ---- Stage z tile: 4 threads per row, 64 contiguous f32 each; exact f32 z2 ----
  {
    const int row = t >> 2;
    const int cbase = (t & 3) * 64;
    const float* zr = z + (size_t)(n0 + row) * D_TOT + cbase;
    float ss = 0.0f;
#pragma unroll
    for (int ii = 0; ii < 16; ++ii) {
      float4 v = reinterpret_cast<const float4*>(zr)[ii];
      float vs[4] = {v.x, v.y, v.z, v.w};
#pragma unroll
      for (int e = 0; e < 4; ++e) {
        float x = vs[e];
        __bf16 h = (__bf16)x;
        __bf16 l = (__bf16)(x - (float)h);
        s_zhi[row * LDS_STRIDE + cbase + ii * 4 + e] = h;
        s_zlo[row * LDS_STRIDE + cbase + ii * 4 + e] = l;
        ss += x * x;
      }
    }
    ss += __shfl_xor(ss, 1);
    ss += __shfl_xor(ss, 2);
    if ((t & 3) == 0) s_zsq[row] = ss;
    if (t < M_TILE) s_rowmin[t] = 0x7F7FFFFFu;
  }
  __syncthreads();

  const int w    = t >> 5;
  const int lane = t & 31;
  const int r    = lane & 15;
  const int h    = lane >> 4;

  float rmin[4][8];
#pragma unroll
  for (int rt = 0; rt < 4; ++rt)
#pragma unroll
    for (int j = 0; j < 8; ++j) rmin[rt][j] = FLT_MAX;

  // ---- Each wave: 4 P-stripes of 16 protos x 4 row-tiles of 16 rows ----
#pragma unroll 1
  for (int i = 0; i < 4; ++i) {
    const int p0 = (w + 8 * i) * 16;
    const float p2v = p2[p0 + r];
    const __bf16* bh_row = phi + (size_t)(p0 + r) * D_TOT;
    const __bf16* bl_row = plo + (size_t)(p0 + r) * D_TOT;

    v8f acc[4] = {};
#pragma unroll 1
    for (int kk = 0; kk < 8; ++kk) {
      const int k0 = kk * 32;
      const int ka = k0 + h * 8;
      // B per-lane layout: same as A, elems 0..7 at h*8, elems 8..15 at 16 + h*8 (L2-resident)
      v16bf Bh = pb_frag16(bh_row + ka);
      v16bf Bl = pb_frag16(bl_row + ka);
#pragma unroll
      for (int rt = 0; rt < 4; ++rt) {
        v16bf Ah = pb_frag16(&s_zhi[(rt * 16 + r) * LDS_STRIDE + ka]);
        v16bf Al = pb_frag16(&s_zlo[(rt * 16 + r) * LDS_STRIDE + ka]);
        acc[rt] = __builtin_amdgcn_wmma_f32_16x16x32_bf16(false, Ah, false, Bh, (short)0, acc[rt], false, false);
        acc[rt] = __builtin_amdgcn_wmma_f32_16x16x32_bf16(false, Al, false, Bh, (short)0, acc[rt], false, false);
        acc[rt] = __builtin_amdgcn_wmma_f32_16x16x32_bf16(false, Ah, false, Bl, (short)0, acc[rt], false, false);
      }
    }

    // Epilogue for this stripe: dist = sqrt(max(z2 + p2 - 2*dot, eps)), update mins.
    float cmin = FLT_MAX;
#pragma unroll
    for (int rt = 0; rt < 4; ++rt) {
#pragma unroll
      for (int j = 0; j < 8; ++j) {
        const int m = rt * 16 + h * 8 + j;  // C layout: vgpr j -> M = j + 8*h
        float sq = s_zsq[m] + p2v - 2.0f * acc[rt][j];
        float d  = sqrtf(fmaxf(sq, 1e-12f));
        rmin[rt][j] = fminf(rmin[rt][j], d);
        cmin = fminf(cmin, d);
      }
    }
    cmin = fminf(cmin, __shfl_xor(cmin, 16));
    if (lane < 16) atomicMin(colmin + p0 + lane, __float_as_uint(cmin));  // dist>=0 -> bit-ordered
  }

  // ---- Row mins: reduce across the 16 lanes of each half-wave, combine waves in LDS ----
#pragma unroll
  for (int rt = 0; rt < 4; ++rt) {
#pragma unroll
    for (int j = 0; j < 8; ++j) {
      float v = rmin[rt][j];
      v = fminf(v, __shfl_xor(v, 1));
      v = fminf(v, __shfl_xor(v, 2));
      v = fminf(v, __shfl_xor(v, 4));
      v = fminf(v, __shfl_xor(v, 8));
      if (r == 0) atomicMin(&s_rowmin[rt * 16 + h * 8 + j], __float_as_uint(v));
    }
  }
  __syncthreads();

  if (w == 0) {
    float v = __uint_as_float(s_rowmin[lane]) + __uint_as_float(s_rowmin[lane + 32]);
    v += __shfl_xor(v, 1);  v += __shfl_xor(v, 2);  v += __shfl_xor(v, 4);
    v += __shfl_xor(v, 8);  v += __shfl_xor(v, 16);
    if (lane == 0) atomicAdd(rowsum, v);
  }
}

__global__ void pb_finalize_kernel(const float* __restrict__ rowsum,
                                   const unsigned* __restrict__ colmin,
                                   float* __restrict__ out) {
  __shared__ float s[8];
  const int t = threadIdx.x;
  float v = __uint_as_float(colmin[t]) + __uint_as_float(colmin[t + 256]);
  v += __shfl_xor(v, 1);  v += __shfl_xor(v, 2);  v += __shfl_xor(v, 4);
  v += __shfl_xor(v, 8);  v += __shfl_xor(v, 16);
  if ((t & 31) == 0) s[t >> 5] = v;
  __syncthreads();
  if (t == 0) {
    float cs = 0.0f;
#pragma unroll
    for (int i = 0; i < 8; ++i) cs += s[i];
    out[0] = 0.05f * (rowsum[0] / (float)N_TOT + cs / (float)P_TOT);
  }
}

extern "C" void kernel_launch(void* const* d_in, const int* in_sizes, int n_in,
                              void* d_out, int out_size, void* d_ws, size_t ws_size,
                              hipStream_t stream) {
  (void)in_sizes; (void)n_in; (void)out_size; (void)ws_size;
  const float* z      = (const float*)d_in[0];
  const float* protos = (const float*)d_in[1];
  char* ws = (char*)d_ws;
  float*    rowsum = (float*)(ws + WS_ROWSUM);
  unsigned* colmin = (unsigned*)(ws + WS_COLMIN);
  float*    p2     = (float*)(ws + WS_P2);
  __bf16*   phi    = (__bf16*)(ws + WS_PHI);
  __bf16*   plo    = (__bf16*)(ws + WS_PLO);

  pb_init_kernel<<<1, 256, 0, stream>>>(rowsum, colmin);
  pb_proto_prep_kernel<<<P_TOT / 8, 256, 0, stream>>>(protos, phi, plo, p2);

  const size_t lds_bytes = (size_t)2 * M_TILE * LDS_STRIDE * 2  // zhi + zlo (bf16)
                         + M_TILE * sizeof(float)               // zsq
                         + M_TILE * sizeof(unsigned);           // rowmin
  pb_dist_kernel<<<N_TOT / M_TILE, 256, lds_bytes, stream>>>(z, phi, plo, p2, rowsum, colmin);
  pb_finalize_kernel<<<1, 256, 0, stream>>>(rowsum, colmin, (float*)d_out);
}
